// Receiver_28080496181522
// MI455X (gfx1250) — compile-verified, matrix operand order checked by reading, not run on hardware
//
#include <hip/hip_runtime.h>
#include <hip/hip_bf16.h>
#include <math.h>

// ---------------------------------------------------------------------------
// Types for CDNA5 WMMA (wave32): v_wmma_f32_16x16x32_bf16
// ---------------------------------------------------------------------------
typedef __attribute__((ext_vector_type(16))) __bf16 bf16x16;
typedef __attribute__((ext_vector_type(8)))  float  v8f;

union FragU { bf16x16 v; uint4 u[2]; };
union BfBits { __bf16 h; unsigned short s; };

// Native f32 -> bf16 conversion (lowers to v_cvt_pk_bf16_f32 on gfx1250)
__device__ __forceinline__ unsigned short f2bf(float f) {
    BfBits u; u.h = (__bf16)f; return u.s;
}

// Load one 16-element bf16 fragment slice for this lane.
// Per ISA 16-bit A layout (16x32): lane<16 owns K {k..k+7, k+16..k+23} of its row;
// rowp points at this lane's row, k = k0 + (lane>=16 ? 8 : 0).
__device__ __forceinline__ bf16x16 load_frag_bf16(const unsigned short* __restrict__ rowp, int k) {
    FragU f;
    f.u[0] = *reinterpret_cast<const uint4*>(rowp + k);
    f.u[1] = *reinterpret_cast<const uint4*>(rowp + k + 16);
    return f.v;
}

// Same fragment, gathering from an f32 row with native bf16 conversion.
__device__ __forceinline__ bf16x16 load_frag_f32(const float* __restrict__ rowp, int k) {
    float4 x0 = *reinterpret_cast<const float4*>(rowp + k);
    float4 x1 = *reinterpret_cast<const float4*>(rowp + k + 4);
    float4 y0 = *reinterpret_cast<const float4*>(rowp + k + 16);
    float4 y1 = *reinterpret_cast<const float4*>(rowp + k + 20);
    bf16x16 v;
    v[0]  = (__bf16)x0.x; v[1]  = (__bf16)x0.y; v[2]  = (__bf16)x0.z; v[3]  = (__bf16)x0.w;
    v[4]  = (__bf16)x1.x; v[5]  = (__bf16)x1.y; v[6]  = (__bf16)x1.z; v[7]  = (__bf16)x1.w;
    v[8]  = (__bf16)y0.x; v[9]  = (__bf16)y0.y; v[10] = (__bf16)y0.z; v[11] = (__bf16)y0.w;
    v[12] = (__bf16)y1.x; v[13] = (__bf16)y1.y; v[14] = (__bf16)y1.z; v[15] = (__bf16)y1.w;
    return v;
}

__device__ __forceinline__ float sigmoidf(float x) { return 1.0f / (1.0f + __expf(-x)); }

#define WMMA_BF16(A, B, C) \
    __builtin_amdgcn_wmma_f32_16x16x32_bf16(false, (A), false, (B), (short)0, (C), false, false)

// ---------------------------------------------------------------------------
// Sizes
// ---------------------------------------------------------------------------
#define VOCAB 50000
#define EMB   256
#define HID   512
#define G4    2048   // 4*HID
#define NFEAT 2048
#define BSZ   128
#define TLEN  64

// ---------------------------------------------------------------------------
// Kernel 1: f32 -> bf16 weight conversion (one-shot, L2-resident afterwards)
// ---------------------------------------------------------------------------
__global__ void __launch_bounds__(256) cvt_bf16_kernel(const float* __restrict__ src,
                                                       unsigned short* __restrict__ dst, int n) {
    int i = blockIdx.x * 256 + threadIdx.x;
    if (i < n) dst[i] = f2bf(src[i]);
}

__global__ void __launch_bounds__(256) init_state_kernel(unsigned short* __restrict__ h,
                                                         float* __restrict__ c, int n) {
    int i = blockIdx.x * 256 + threadIdx.x;
    if (i < n) { h[i] = 0; c[i] = 0.0f; }
}

// ---------------------------------------------------------------------------
// Kernel 2: fused embedding-gather + input-projection GEMM
//   x_proj[t*B + b, g] = sum_e emb_table[m[b,t], e] * W_ih[g, e] + b_ih[g]
//   M=8192 (=T*B), N=2048, K=256. One wave = 16 rows x 128 cols (A reused 8x).
//   8192 waves -> occupancy hides L2 latency; no manual pipelining needed.
// ---------------------------------------------------------------------------
__global__ void __launch_bounds__(256)
xproj_kernel(const int* __restrict__ m, const float* __restrict__ emb,
             const unsigned short* __restrict__ Wih, const float* __restrict__ b_ih,
             float* __restrict__ xproj) {
    const int lane = threadIdx.x & 31;
    const int wave = threadIdx.x >> 5;
    const int gid  = blockIdx.x * 8 + wave;   // 8192 waves total
    const int mt   = gid >> 4;                // 512 row-tiles of 16
    const int ns   = gid & 15;                // 16 col-strips of 128
    const int row0 = mt * 16;
    const int n0   = ns * 128;
    const int lc   = lane & 15;
    const int half = lane >> 4;
    const int koff = half * 8;

    // gather pointer: this lane's A row (row index = t*128 + b)
    const int arow = row0 + lc;
    const int tt   = arow >> 7;       // / BSZ
    const int bb   = arow & 127;      // % BSZ
    const int tok  = m[bb * TLEN + tt];
    const float* arowp = emb + (size_t)tok * EMB;

    v8f zero = {};
    v8f acc[8];
#pragma unroll
    for (int j = 0; j < 8; ++j) acc[j] = zero;

#pragma unroll
    for (int kk = 0; kk < 8; ++kk) {            // K = 256 = 8 * 32
        const int k = kk * 32 + koff;
        bf16x16 a = load_frag_f32(arowp, k);
#pragma unroll
        for (int j = 0; j < 8; ++j) {
            const unsigned short* brow = Wih + (size_t)(n0 + j * 16 + lc) * EMB;
            bf16x16 b = load_frag_bf16(brow, k);
            acc[j] = WMMA_BF16(a, b, acc[j]);
        }
    }

#pragma unroll
    for (int j = 0; j < 8; ++j) {
        const int col  = n0 + j * 16 + lc;
        const float bi = b_ih[col];
#pragma unroll
        for (int i = 0; i < 8; ++i) {
            const int row = row0 + half * 8 + i;
            xproj[(size_t)row * G4 + col] = acc[j][i] + bi;
        }
    }
}

// ---------------------------------------------------------------------------
// Kernel 3: one LSTM step (launched 64x, sequentially on stream)
//   gates = xproj[t] + h @ W_hh^T + b_hh ; pointwise LSTM update.
//   Each wave: one 16x16 tile of h-columns, all FOUR gates (N offsets
//   0/512/1024/1536) -> update is wave-local, no inter-block sync needed.
//   Only 256 waves run chip-wide (~1 per SIMD): L2 latency cannot be hidden
//   by other waves, so fragments are software-pipelined with prefetch
//   distance PD=4 (loads for k+4 in flight while WMMAs consume k).
//   h ping-pongs between two bf16 buffers across launches.
// ---------------------------------------------------------------------------
#define STEP_PD 4

__global__ void __launch_bounds__(128)
lstm_step_kernel(const float* __restrict__ xproj_t, const unsigned short* __restrict__ Whh,
                 const float* __restrict__ b_hh, const unsigned short* __restrict__ h_in,
                 float* __restrict__ c, unsigned short* __restrict__ h_out) {
    const int lane = threadIdx.x & 31;
    const int wave = threadIdx.x >> 5;
    const int tile = blockIdx.x * 4 + wave;   // 256 tiles = 8 (B/16) x 32 (H/16)
    const int m0   = (tile >> 5) * 16;
    const int n0   = (tile & 31) * 16;
    const int lc   = lane & 15;
    const int half = lane >> 4;
    const int koff = half * 8;

    const unsigned short* arowp = h_in + (size_t)(m0 + lc) * HID;
    const unsigned short* brow0 = Whh + (size_t)(0 * HID + n0 + lc) * HID;
    const unsigned short* brow1 = Whh + (size_t)(1 * HID + n0 + lc) * HID;
    const unsigned short* brow2 = Whh + (size_t)(2 * HID + n0 + lc) * HID;
    const unsigned short* brow3 = Whh + (size_t)(3 * HID + n0 + lc) * HID;

    v8f zero = {};
    v8f acc[4];
#pragma unroll
    for (int q = 0; q < 4; ++q) acc[q] = zero;

    bf16x16 aF[16];
    bf16x16 bF[16][4];

    // pipeline prologue: stages 0..PD-1 in flight
#pragma unroll
    for (int kk = 0; kk < STEP_PD; ++kk) {
        const int k = kk * 32 + koff;
        aF[kk]    = load_frag_bf16(arowp, k);
        bF[kk][0] = load_frag_bf16(brow0, k);
        bF[kk][1] = load_frag_bf16(brow1, k);
        bF[kk][2] = load_frag_bf16(brow2, k);
        bF[kk][3] = load_frag_bf16(brow3, k);
    }

#pragma unroll
    for (int kk = 0; kk < 16; ++kk) {           // K = 512 = 16 * 32
        if (kk + STEP_PD < 16) {                // issue loads PD iterations ahead
            const int k = (kk + STEP_PD) * 32 + koff;
            aF[kk + STEP_PD]    = load_frag_bf16(arowp, k);
            bF[kk + STEP_PD][0] = load_frag_bf16(brow0, k);
            bF[kk + STEP_PD][1] = load_frag_bf16(brow1, k);
            bF[kk + STEP_PD][2] = load_frag_bf16(brow2, k);
            bF[kk + STEP_PD][3] = load_frag_bf16(brow3, k);
        }
#pragma unroll
        for (int q = 0; q < 4; ++q)
            acc[q] = WMMA_BF16(aF[kk], bF[kk][q], acc[q]);
    }

    const int col = n0 + lc;
    const float bi = b_hh[col];
    const float bf = b_hh[col + HID];
    const float bg = b_hh[col + 2 * HID];
    const float bo = b_hh[col + 3 * HID];

#pragma unroll
    for (int i = 0; i < 8; ++i) {
        const int row = m0 + half * 8 + i;               // batch index b
        const float* xr = xproj_t + (size_t)row * G4;
        float gi = acc[0][i] + xr[col]           + bi;
        float gf = acc[1][i] + xr[col + HID]     + bf;
        float gg = acc[2][i] + xr[col + 2 * HID] + bg;
        float go = acc[3][i] + xr[col + 3 * HID] + bo;
        const size_t idx = (size_t)row * HID + col;
        float cn = sigmoidf(gf) * c[idx] + sigmoidf(gi) * tanhf(gg);
        c[idx] = cn;
        h_out[idx] = f2bf(sigmoidf(go) * tanhf(cn));
    }
}

// ---------------------------------------------------------------------------
// Kernel 4: output GEMM  out = h @ W_out^T + b_out   (M=128,N=2048,K=512)
//   Also low-occupancy (1024 waves) -> same PD=4 fragment pipelining.
// ---------------------------------------------------------------------------
__global__ void __launch_bounds__(256)
out_kernel(const unsigned short* __restrict__ h, const unsigned short* __restrict__ Wout,
           const float* __restrict__ b_out, float* __restrict__ out) {
    const int lane = threadIdx.x & 31;
    const int wave = threadIdx.x >> 5;
    const int tile = blockIdx.x * 8 + wave;   // 1024 tiles = 8 x 128
    const int m0   = (tile >> 7) * 16;
    const int n0   = (tile & 127) * 16;
    const int lc   = lane & 15;
    const int half = lane >> 4;
    const int koff = half * 8;

    const unsigned short* arowp = h + (size_t)(m0 + lc) * HID;
    const unsigned short* browp = Wout + (size_t)(n0 + lc) * HID;

    v8f acc = {};
    bf16x16 aF[16];
    bf16x16 bF[16];

#pragma unroll
    for (int kk = 0; kk < STEP_PD; ++kk) {
        const int k = kk * 32 + koff;
        aF[kk] = load_frag_bf16(arowp, k);
        bF[kk] = load_frag_bf16(browp, k);
    }

#pragma unroll
    for (int kk = 0; kk < 16; ++kk) {
        if (kk + STEP_PD < 16) {
            const int k = (kk + STEP_PD) * 32 + koff;
            aF[kk + STEP_PD] = load_frag_bf16(arowp, k);
            bF[kk + STEP_PD] = load_frag_bf16(browp, k);
        }
        acc = WMMA_BF16(aF[kk], bF[kk], acc);
    }

    const int col  = n0 + lc;
    const float bb = b_out[col];
#pragma unroll
    for (int i = 0; i < 8; ++i) {
        const int row = m0 + half * 8 + i;
        out[(size_t)row * NFEAT + col] = acc[i] + bb;
    }
}

// ---------------------------------------------------------------------------
// Host-side launch. Workspace layout (256B aligned):
//   [0)        W_ih  bf16  2048x256  (1,048,576 B)
//   [1 MiB)    W_hh  bf16  2048x512  (2,097,152 B)
//   [3 MiB)    W_out bf16  2048x512  (2,097,152 B)
//   [...]      h_a / h_b   bf16 128x512 (131,072 B each)
//   [...]      c     f32   128x512  (262,144 B)
//   [...]      x_proj f32  8192x2048 (67,108,864 B)     total ~73 MB
// ---------------------------------------------------------------------------
extern "C" void kernel_launch(void* const* d_in, const int* in_sizes, int n_in,
                              void* d_out, int out_size, void* d_ws, size_t ws_size,
                              hipStream_t stream) {
    (void)in_sizes; (void)n_in; (void)out_size; (void)ws_size;

    const int*   m     = (const int*)  d_in[0];
    // d_in[1] = images (unused by the reference forward)
    const float* emb   = (const float*)d_in[2];
    const float* W_ih  = (const float*)d_in[3];
    const float* W_hh  = (const float*)d_in[4];
    const float* b_ih  = (const float*)d_in[5];
    const float* b_hh  = (const float*)d_in[6];
    const float* W_out = (const float*)d_in[7];
    const float* b_out = (const float*)d_in[8];
    float*       out   = (float*)d_out;

    char* w = (char*)d_ws;
    unsigned short* Wih_bf  = (unsigned short*)(w + 0);
    unsigned short* Whh_bf  = (unsigned short*)(w + 1048576);
    unsigned short* Wout_bf = (unsigned short*)(w + 3145728);
    unsigned short* h_a     = (unsigned short*)(w + 5242880);
    unsigned short* h_b     = (unsigned short*)(w + 5373952);
    float*          c       = (float*)         (w + 5505024);
    float*          xproj   = (float*)         (w + 5767168);

    // 1) weight conversion (tiny; results live in L2 for the whole run)
    cvt_bf16_kernel<<<(G4 * EMB + 255) / 256, 256, 0, stream>>>(W_ih,  Wih_bf,  G4 * EMB);
    cvt_bf16_kernel<<<(G4 * HID + 255) / 256, 256, 0, stream>>>(W_hh,  Whh_bf,  G4 * HID);
    cvt_bf16_kernel<<<(G4 * HID + 255) / 256, 256, 0, stream>>>(W_out, Wout_bf, G4 * HID);
    init_state_kernel<<<(BSZ * HID + 255) / 256, 256, 0, stream>>>(h_a, c, BSZ * HID);

    // 2) fused gather + big input-projection GEMM (8192x2048x256, bf16 WMMA)
    xproj_kernel<<<1024, 256, 0, stream>>>(m, emb, Wih_bf, b_ih, xproj);

    // 3) 64 sequential LSTM steps; h ping-pongs between bf16 buffers
    for (int t = 0; t < TLEN; ++t) {
        const unsigned short* hin  = (t & 1) ? h_b : h_a;
        unsigned short*       hout = (t & 1) ? h_a : h_b;
        lstm_step_kernel<<<64, 128, 0, stream>>>(xproj + (size_t)t * BSZ * G4,
                                                 Whh_bf, b_hh, hin, c, hout);
    }

    // after t=63 (odd) the final h lives in h_a
    out_kernel<<<128, 256, 0, stream>>>(h_a, Wout_bf, b_out, out);
}
